// GAT_22127671509494
// MI455X (gfx1250) — matrix operand order repaired and checked by reading, not compiled, not run on hardware
//
#include <hip/hip_runtime.h>

// ---------------------------------------------------------------------------
// GAT forward on MI455X (gfx1250, wave32).
// Heavy GEMMs (x@W, attn@Wh) on v_wmma_f32_16x16x32_bf16; flash-style online
// softmax for the masked attention; adj (64MB) + intermediates stay L2-resident.
// ---------------------------------------------------------------------------

typedef __bf16 bf16;
typedef __attribute__((ext_vector_type(16))) __bf16 v16bf;
typedef __attribute__((ext_vector_type(8)))  float  v8f;

#define NN   4096
#define FIN  512
#define FH   64
#define NH   8
#define NC   40
#define NCP  48     // classes padded to 3 WMMA N-tiles

struct alignas(16) Q4 { unsigned u0, u1, u2, u3; };
struct alignas(16) F4 { float x, y, z, w; };

__device__ __forceinline__ bf16 f2bf(float f) {
  union { float f; unsigned u; } c; c.f = f;
  unsigned r = c.u + 0x7fffu + ((c.u >> 16) & 1u);   // round-to-nearest-even
  union { unsigned short s; bf16 b; } o; o.s = (unsigned short)(r >> 16);
  return o.b;
}
__device__ __forceinline__ float bf2f(bf16 b) {
  union { unsigned short s; bf16 b; } i; i.b = b;
  union { float f; unsigned u; } o; o.u = ((unsigned)i.s) << 16;
  return o.f;
}

// -------------------------------- prep -------------------------------------
__global__ void k_cvt_bf16(const float* __restrict__ src, bf16* __restrict__ dst, int n) {
  int i = blockIdx.x * blockDim.x + threadIdx.x;
  if (i < n) dst[i] = f2bf(src[i]);
}

// WbT[h][n][k] = W_heads[h][k][n]   (K-contiguous for WMMA B fills)
__global__ void k_prep_w1(const float* __restrict__ W, bf16* __restrict__ WT) {
  int i = blockIdx.x * blockDim.x + threadIdx.x;
  if (i >= NH * FH * FIN) return;
  int h = i / (FH * FIN); int rem = i - h * FH * FIN;
  int n = rem / FIN;      int k = rem - n * FIN;
  WT[i] = f2bf(W[((size_t)h * FIN + k) * FH + n]);
}

// WobT[c][k] = W_out[k][c], zero-padded to NCP rows
__global__ void k_prep_w2(const float* __restrict__ W, bf16* __restrict__ WT) {
  int i = blockIdx.x * blockDim.x + threadIdx.x;
  if (i >= NCP * FIN) return;
  int c = i / FIN; int k = i - c * FIN;
  WT[i] = (c < NC) ? f2bf(W[(size_t)k * NC + c]) : f2bf(0.0f);
}

// ------------------------------ projection ---------------------------------
// Out[m][n] = sum_k X[m][k] * W[k][n], stored transposed bf16: OutT[n][m].
// block = 32 * nTilesN (wave w -> N-tile w); grid = (NN/16, heads).
__global__ void k_proj(const bf16* __restrict__ X, const bf16* __restrict__ WT,
                       bf16* __restrict__ OutT, int Nout) {
  int lane = threadIdx.x & 31;
  int wv   = threadIdx.x >> 5;
  int half = lane >> 4;
  int l16  = lane & 15;
  int m0   = blockIdx.x * 16;
  int head = blockIdx.y;
  int n0   = wv * 16;

  const bf16* xp = X + (size_t)(m0 + l16) * FIN;
  const bf16* wt = WT + ((size_t)head * Nout + n0 + l16) * FIN;

  v8f acc = {0.f,0.f,0.f,0.f,0.f,0.f,0.f,0.f};
  union { v16bf v; Q4 q[2]; } A, B;
  for (int k0 = 0; k0 < FIN; k0 += 32) {
    // A 16x32 bf16: lane l holds row l%16, K = 16*(e/8) + 8*(l/16) + e%8
    A.q[0] = *(const Q4*)(xp + k0 + 8 * half);
    A.q[1] = *(const Q4*)(xp + k0 + 8 * half + 16);
    // B 32x16 bf16: lane l holds col l%16, K = 16*(e/8) + 8*(l/16) + e%8
    B.q[0] = *(const Q4*)(wt + k0 + 8 * half);
    B.q[1] = *(const Q4*)(wt + k0 + 8 * half + 16);
    acc = __builtin_amdgcn_wmma_f32_16x16x32_bf16(false, A.v, false, B.v,
                                                  (short)0, acc, false, false);
  }
  // C/D: lane l, VGPR r -> row 8*(l/16)+r, col l%16. 8 rows at fixed n are
  // contiguous in OutT[n][m] -> single 16B store.
  union { Q4 q; bf16 e[8]; } st;
#pragma unroll
  for (int r = 0; r < 8; r++) st.e[r] = f2bf(acc[r]);
  *(Q4*)(OutT + ((size_t)head * Nout + n0 + l16) * NN + m0 + 8 * half) = st.q;
}

// -------------------------- attention scores f1/f2 -------------------------
__global__ void k_fscore(const bf16* __restrict__ WhT, const float* __restrict__ a,
                         float* __restrict__ f1, float* __restrict__ f2,
                         int Nout, int Fo) {
  int m = blockIdx.x * blockDim.x + threadIdx.x;
  int h = blockIdx.y;
  if (m >= NN) return;
  const bf16*  base = WhT + (size_t)h * Nout * NN + m;
  const float* ap   = a + h * 2 * Fo;
  float s1 = 0.f, s2 = 0.f;
  for (int n = 0; n < Fo; n++) {
    float w = bf2f(base[(size_t)n * NN]);
    s1 += w * ap[n];
    s2 += w * ap[Fo + n];
  }
  f1[h * NN + m] = s1;
  f2[h * NN + m] = s2;
}

// --------------------- masked attention + aggregation ----------------------
// Flash-style: per wave 16 query rows; j-blocks of 32; P (16x32 bf16) built in
// registers in exact WMMA A layout; V (32 x 16*NT bf16) staged in LDS.
template <int NT>
__global__ void k_attn(const float* __restrict__ adj,
                       const bf16* __restrict__ WhT,   // [heads][16*NT][NN]
                       const float* __restrict__ f1g,
                       const float* __restrict__ f2g,
                       bf16*  __restrict__ outB,       // layer1: [NN][FIN]
                       float* __restrict__ outF) {     // layer2: [NN][NCP]
  constexpr int NOUT = 16 * NT;
  constexpr int VS   = 40;                             // padded LDS row stride
  __shared__ __attribute__((aligned(16))) bf16 Vs[NOUT * VS];

  int tid  = threadIdx.x;
  int lane = tid & 31, wv = tid >> 5;
  int half = lane >> 4, l16 = lane & 15;
  int head = blockIdx.y;
  int m0   = blockIdx.x * 128 + wv * 16;
  int myrow = m0 + l16;

  const float* adjrow = adj + (size_t)myrow * NN;
  const float* f2p    = f2g + head * NN;
  const bf16*  wht    = WhT + (size_t)head * NOUT * NN;
  float f1v = f1g[head * NN + myrow];

  float mrow = -9e15f, lsum = 0.f;
  v8f zero = {0.f,0.f,0.f,0.f,0.f,0.f,0.f,0.f};
  v8f acc[NT];
#pragma unroll
  for (int t = 0; t < NT; t++) acc[t] = zero;

  for (int j0 = 0; j0 < NN; j0 += 32) {
    __syncthreads();
    if (tid < NOUT * 4) {           // cooperative V-tile load: 16B per thread
      int n = tid >> 2, kq = (tid & 3) * 8;
      *(Q4*)&Vs[n * VS + kq] = *(const Q4*)(wht + (size_t)n * NN + j0 + kq);
    }
    __syncthreads();

    // scores for this lane's 16 elements: j = j0 + 16*(e/8) + 8*half + e%8
    int jb = j0 + 8 * half;
    F4 a0 = *(const F4*)(adjrow + jb);      F4 a1 = *(const F4*)(adjrow + jb + 4);
    F4 a2 = *(const F4*)(adjrow + jb + 16); F4 a3 = *(const F4*)(adjrow + jb + 20);
    F4 g0 = *(const F4*)(f2p + jb);         F4 g1 = *(const F4*)(f2p + jb + 4);
    F4 g2 = *(const F4*)(f2p + jb + 16);    F4 g3 = *(const F4*)(f2p + jb + 20);
    float av[16] = {a0.x,a0.y,a0.z,a0.w, a1.x,a1.y,a1.z,a1.w,
                    a2.x,a2.y,a2.z,a2.w, a3.x,a3.y,a3.z,a3.w};
    float gv[16] = {g0.x,g0.y,g0.z,g0.w, g1.x,g1.y,g1.z,g1.w,
                    g2.x,g2.y,g2.z,g2.w, g3.x,g3.y,g3.z,g3.w};
    float sv[16];
#pragma unroll
    for (int e = 0; e < 16; e++) {
      float s = f1v + gv[e];
      s = (s > 0.f) ? s : 0.2f * s;               // leaky relu
      sv[e] = (av[e] > 0.f) ? s : -9e15f;         // adjacency mask
    }

    // online softmax (rows live in lanes l%16; halves combined via xor-16)
    float bmax = sv[0];
#pragma unroll
    for (int e = 1; e < 16; e++) bmax = fmaxf(bmax, sv[e]);
    bmax = fmaxf(bmax, __shfl_xor(bmax, 16));
    float mnew  = fmaxf(mrow, bmax);
    float scale = __expf(mrow - mnew);

    union { v16bf v; bf16 e[16]; } A;
    float psum = 0.f;
#pragma unroll
    for (int e = 0; e < 16; e++) {
      float pv = __expf(sv[e] - mnew);
      psum += pv;
      A.e[e] = f2bf(pv);
    }
    psum += __shfl_xor(psum, 16);
    lsum = lsum * scale + psum;
    mrow = mnew;

    // rescale accumulators per row (acc row = 8*half + r, owned by lane row)
#pragma unroll
    for (int r = 0; r < 8; r++) {
      float fct = __shfl(scale, 8 * half + r);
#pragma unroll
      for (int t = 0; t < NT; t++) acc[t][r] *= fct;
    }

    // acc += P @ V ; B operand from LDS (lane l: col l%16, K = 16*(e/8) + 8*half + e%8)
#pragma unroll
    for (int t = 0; t < NT; t++) {
      union { v16bf v; Q4 q[2]; } B;
      const bf16* bp = &Vs[(16 * t + l16) * VS + 8 * half];
      B.q[0] = *(const Q4*)bp;
      B.q[1] = *(const Q4*)(bp + 16);
      acc[t] = __builtin_amdgcn_wmma_f32_16x16x32_bf16(false, A.v, false, B.v,
                                                       (short)0, acc[t], false, false);
    }
  }

  // normalize + ELU, write out
#pragma unroll
  for (int r = 0; r < 8; r++) {
    float ls  = __shfl(lsum, 8 * half + r);
    float inv = 1.f / ls;
    int row = m0 + 8 * half + r;
#pragma unroll
    for (int t = 0; t < NT; t++) {
      float v = acc[t][r] * inv;
      v = (v > 0.f) ? v : (__expf(v) - 1.f);      // elu
      int col = 16 * t + l16;
      if (outB) {
        outB[(size_t)row * FIN + head * FH + col] = f2bf(v);   // concat heads
      } else if (col < NC) {
        outF[(size_t)row * NCP + col] = v;
      }
    }
  }
}

// ------------------------------ log-softmax --------------------------------
__global__ void k_logsoftmax(const float* __restrict__ in, float* __restrict__ out) {
  int m = blockIdx.x * blockDim.x + threadIdx.x;
  if (m >= NN) return;
  const float* r = in + (size_t)m * NCP;
  float v[NC];
  float mx = -3.4e38f;
  for (int c = 0; c < NC; c++) { v[c] = r[c]; mx = fmaxf(mx, v[c]); }
  float se = 0.f;
  for (int c = 0; c < NC; c++) se += __expf(v[c] - mx);
  float ls = mx + __logf(se);
  for (int c = 0; c < NC; c++) out[(size_t)m * NC + c] = v[c] - ls;
}

// ------------------------------- launcher ----------------------------------
extern "C" void kernel_launch(void* const* d_in, const int* in_sizes, int n_in,
                              void* d_out, int out_size, void* d_ws, size_t ws_size,
                              hipStream_t stream) {
  const float* x   = (const float*)d_in[0];
  const float* adj = (const float*)d_in[1];
  const float* Wh  = (const float*)d_in[2];
  const float* ah  = (const float*)d_in[3];
  const float* Wo  = (const float*)d_in[4];
  const float* ao  = (const float*)d_in[5];
  float* out = (float*)d_out;

  char* p = (char*)d_ws;
  auto alloc = [&](size_t bytes) -> char* {
    char* r = p; p += (bytes + 255) & ~((size_t)255); return r;
  };
  bf16*  xb   = (bf16*) alloc((size_t)NN * FIN * 2);
  bf16*  WbT  = (bf16*) alloc((size_t)NH * FH * FIN * 2);
  bf16*  WobT = (bf16*) alloc((size_t)NCP * FIN * 2);
  bf16*  WhbT = (bf16*) alloc((size_t)NH * FH * NN * 2);
  float* f1   = (float*)alloc((size_t)NH * NN * 4);
  float* f2   = (float*)alloc((size_t)NH * NN * 4);
  bf16*  hb   = (bf16*) alloc((size_t)NN * FIN * 2);
  bf16*  W2T  = (bf16*) alloc((size_t)NCP * NN * 2);
  float* f1o  = (float*)alloc((size_t)NN * 4);
  float* f2o  = (float*)alloc((size_t)NN * 4);
  float* out2 = (float*)alloc((size_t)NN * NCP * 4);

  k_cvt_bf16<<<(NN * FIN + 255) / 256, 256, 0, stream>>>(x, xb, NN * FIN);
  k_prep_w1<<<(NH * FH * FIN + 255) / 256, 256, 0, stream>>>(Wh, WbT);
  k_prep_w2<<<(NCP * FIN + 255) / 256, 256, 0, stream>>>(Wo, WobT);

  // layer 1
  k_proj<<<dim3(NN / 16, NH), 128, 0, stream>>>(xb, WbT, WhbT, FH);
  k_fscore<<<dim3(NN / 256, NH), 256, 0, stream>>>(WhbT, ah, f1, f2, FH, FH);
  k_attn<4><<<dim3(NN / 128, NH), 256, 0, stream>>>(adj, WhbT, f1, f2, hb, nullptr);

  // layer 2
  k_proj<<<dim3(NN / 16, 1), 96, 0, stream>>>(hb, WobT, W2T, NCP);
  k_fscore<<<dim3(NN / 256, 1), 256, 0, stream>>>(W2T, ao, f1o, f2o, NCP, NC);
  k_attn<3><<<dim3(NN / 128, 1), 256, 0, stream>>>(adj, W2T, f1o, f2o, nullptr, out2);

  k_logsoftmax<<<(NN + 255) / 256, 256, 0, stream>>>(out2, out);
}
